// LaneAwareAttention_16896401343260
// MI455X (gfx1250) — matrix operand order repaired and checked by reading, not compiled, not run on hardware
//
#include <hip/hip_runtime.h>
#include <hip/hip_bf16.h>

typedef _Float16 half_t;
typedef __attribute__((ext_vector_type(16))) _Float16 v16h;
typedef __attribute__((ext_vector_type(8)))  float    v8f;

#define B_    16
#define N_    400
#define C_    512
#define NH_   8
#define HD_   64
#define NT_   25      // N_/16
#define PADN  416     // N_ padded to multiple of 32 (and vt row stride)

#define WMMA_F16(a, b, c) \
  __builtin_amdgcn_wmma_f32_16x16x32_f16(false, (a), false, (b), (short)0, (c), false, false)

// ---------------------------------------------------------------------------
// Fragment loaders (CDNA5 wave32 VGPR layouts, CDNA5 ISA §7.12.2)
// ---------------------------------------------------------------------------

// A-matrix 16x32 f16, row-major source with leading dim `ld` (elements).
// lane<16: row=lane, elems 0..7 = K0..7, elems 8..15 = K16..23
// lane>=16: row=lane-16, elems 0..7 = K8..15, elems 8..15 = K24..31
static __device__ __forceinline__ v16h load_a_frag(const half_t* src, int ld) {
  const int lane = threadIdx.x & 31;
  const int row  = lane & 15;
  const int khi  = (lane >> 4) & 1;
  const half_t* p = src + row * ld + khi * 8;
  v16h a;
#pragma unroll
  for (int i = 0; i < 8; ++i) a[i] = p[i];
#pragma unroll
  for (int i = 0; i < 8; ++i) a[8 + i] = p[16 + i];
  return a;
}

// B-matrix 32x16 (B = M^T) loading from row-major M: src -> M[n0][k0], ld = row
// stride of M. lane<16: col=lane, elems 0..7 = K0..7, 8..15 = K16..23; lane>=16: K8..15, K24..31.
static __device__ __forceinline__ v16h load_b_frag(const half_t* src, int ld) {
  const int lane = threadIdx.x & 31;
  const int col  = lane & 15;
  const int khi  = (lane >> 4) & 1;
  const half_t* p = src + col * ld + khi * 8;
  v16h b;
#pragma unroll
  for (int i = 0; i < 8; ++i) b[i] = p[i];
#pragma unroll
  for (int i = 0; i < 8; ++i) b[8 + i] = p[16 + i];
  return b;
}

// ---------------------------------------------------------------------------
// Kernel 0: fp32 -> fp16 convert (grid-stride)
// ---------------------------------------------------------------------------
__global__ void cvt_f32_to_f16(const float* __restrict__ in, half_t* __restrict__ out, int n) {
  for (int i = blockIdx.x * blockDim.x + threadIdx.x; i < n; i += gridDim.x * blockDim.x)
    out[i] = (half_t)in[i];
}

// ---------------------------------------------------------------------------
// Kernel 0b: zero the pad columns (m = 400..415) of transposed V so padded
// attention probs multiply exact zeros (WMMA propagates NaN even vs 0).
// ---------------------------------------------------------------------------
__global__ void zero_vt_pad(half_t* __restrict__ vt) {
  const int total = B_ * NH_ * HD_ * (PADN - N_);   // 131072
  for (int i = blockIdx.x * blockDim.x + threadIdx.x; i < total;
       i += gridDim.x * blockDim.x) {
    const int row = i / (PADN - N_);
    const int m   = N_ + i % (PADN - N_);
    vt[(long)row * PADN + m] = (half_t)0.0f;
  }
}

// ---------------------------------------------------------------------------
// Kernel 1: QKV GEMM  Y = Xh * Wqkv^T + b.
//   q -> [B*NH, N, HD] (scaled by 1/sqrt(hd)),  k -> [B*NH, N, HD],
//   v -> TRANSPOSED [B*NH, HD, PADN]  (so P*V B-frags are contiguous loads).
// One wave computes a 16x64 tile. grid = (B*N/16, 3C/64) = (400, 24), block=32.
// ---------------------------------------------------------------------------
__global__ void qkv_gemm(const half_t* __restrict__ X, const half_t* __restrict__ W,
                         const float* __restrict__ bias,
                         half_t* __restrict__ Q, half_t* __restrict__ K,
                         half_t* __restrict__ VT) {
  const int r0 = blockIdx.x * 16;        // row tile in [0, B*N)
  const int c0 = blockIdx.y * 64;        // col tile in [0, 3C)

  v8f acc[4] = {};
  for (int k0 = 0; k0 < C_; k0 += 32) {
    v16h a = load_a_frag(X + (long)r0 * C_ + k0, C_);
#pragma unroll
    for (int t = 0; t < 4; ++t) {
      v16h b = load_b_frag(W + (long)(c0 + t * 16) * C_ + k0, C_);
      acc[t] = WMMA_F16(a, b, acc[t]);
    }
  }

  const int lane  = threadIdx.x & 31;
  const int coll  = lane & 15;
  const int rbase = (lane >> 4) * 8;
#pragma unroll
  for (int t = 0; t < 4; ++t) {
    const int c  = c0 + t * 16 + coll;
    const int s  = c / C_;               // 0=q 1=k 2=v (tile never straddles)
    const int cc = c % C_;
    const int h  = cc / HD_;
    const int d  = cc % HD_;
    const float sc = (s == 0) ? 0.125f : 1.0f;   // hd^-0.5 = 1/8
    const float bb = bias[c];
#pragma unroll
    for (int r = 0; r < 8; ++r) {
      const int row = r0 + rbase + r;
      const int b_  = row / N_;
      const int n   = row % N_;
      const long bh = (long)b_ * NH_ + h;
      const half_t val = (half_t)((acc[t][r] + bb) * sc);
      if (s == 0)      Q[(bh * N_ + n) * HD_ + d] = val;
      else if (s == 1) K[(bh * N_ + n) * HD_ + d] = val;
      else             VT[(bh * HD_ + d) * PADN + n] = val;   // transposed
    }
  }
}

// ---------------------------------------------------------------------------
// Kernel 2: attention. One wave per (b, h, 16-row q tile). grid = 3200, block=32.
//   S = q*k^T (+rel-pos bias) -> LDS fp32 -> 2-lanes-per-row softmax ->
//   f16 probs zero-padded to 416 -> O = P*V (V transposed, contiguous frags)
//   -> f16 into token-major [B*N, C] buffer for proj.
// ---------------------------------------------------------------------------
__global__ void attn_kernel(const half_t* __restrict__ Q, const half_t* __restrict__ K,
                            const half_t* __restrict__ VT,
                            const float* __restrict__ relh, const float* __restrict__ relw,
                            half_t* __restrict__ Aout) {
  __shared__ float  sS[16 * PADN];
  __shared__ half_t sP[16 * PADN];
  __shared__ float  sRH[39 * NH_];
  __shared__ float  sRW[39 * NH_];

  const int qt = blockIdx.x % NT_;
  const int bh = blockIdx.x / NT_;       // b*NH + h
  const int h  = bh % NH_;
  const int b  = bh / NH_;
  const int i0 = qt * 16;

  const half_t* qp  = Q + ((long)bh * N_ + i0) * HD_;
  const half_t* kp  = K + (long)bh * N_ * HD_;
  const half_t* vtp = VT + (long)bh * HD_ * PADN;

  const int lane  = threadIdx.x & 31;
  const int coll  = lane & 15;
  const int rbase = (lane >> 4) * 8;

  // stage rel-pos tables into LDS (39*8 floats each)
  for (int i = lane; i < 39 * NH_; i += 32) { sRH[i] = relh[i]; sRW[i] = relw[i]; }

  // q tile fragments (K = 0..31 and 32..63); scale already folded into Q.
  const v16h a0 = load_a_frag(qp + 0, HD_);
  const v16h a1 = load_a_frag(qp + 32, HD_);
  __syncthreads();

  // ---- scores + rel-pos bias into LDS ----
  for (int j = 0; j < NT_; ++j) {
    const int m0 = j * 16;
    v16h b0 = load_b_frag(kp + (long)m0 * HD_ + 0, HD_);
    v16h b1 = load_b_frag(kp + (long)m0 * HD_ + 32, HD_);
    v8f c = {};
    c = WMMA_F16(a0, b0, c);
    c = WMMA_F16(a1, b1, c);

    const int m  = m0 + coll;
    const int mh = m / 20, mw = m % 20;
#pragma unroll
    for (int r = 0; r < 8; ++r) {
      const int i  = i0 + rbase + r;
      const int ih = i / 20, iw = i % 20;
      const float bias = sRH[(ih - mh + 19) * NH_ + h] + sRW[(iw - mw + 19) * NH_ + h];
      sS[(rbase + r) * PADN + m] = c[r] + bias;
    }
  }
  __syncthreads();

  // ---- softmax: 2 lanes per row, 200 columns each; combine via shfl_xor(16) ----
  {
    const int row  = lane & 15;
    const int half = lane >> 4;
    const int m0   = half * 200;
    float* srow  = sS + row * PADN;
    half_t* prow = sP + row * PADN;

    float mx = -3.0e38f;
    for (int m = m0; m < m0 + 200; ++m) mx = fmaxf(mx, srow[m]);
    mx = fmaxf(mx, __shfl_xor(mx, 16));

    float sum = 0.0f;
    for (int m = m0; m < m0 + 200; ++m) {
      const float e = __expf(srow[m] - mx);
      srow[m] = e;
      sum += e;
    }
    sum += __shfl_xor(sum, 16);
    const float inv = 1.0f / sum;

    for (int m = m0; m < m0 + 200; ++m) prow[m] = (half_t)(srow[m] * inv);
    if (half) for (int m = N_; m < PADN; ++m) prow[m] = (half_t)0.0f;
  }
  __syncthreads();

  // ---- O = P * V  (16 x 64, K = 416 in 13 steps; V transposed -> contiguous) ----
  v8f o[4] = {};
  for (int kk = 0; kk < PADN; kk += 32) {
    v16h a = load_a_frag((const half_t*)sP + kk, PADN);
#pragma unroll
    for (int t = 0; t < 4; ++t) {
      v16h bf = load_b_frag(vtp + (long)(t * 16) * PADN + kk, PADN);
      o[t] = WMMA_F16(a, bf, o[t]);
    }
  }

#pragma unroll
  for (int t = 0; t < 4; ++t)
#pragma unroll
    for (int r = 0; r < 8; ++r) {
      const int i = i0 + rbase + r;
      Aout[((long)b * N_ + i) * C_ + h * HD_ + t * 16 + coll] = (half_t)o[t][r];
    }
}

// ---------------------------------------------------------------------------
// Kernel 3: output projection  out = Ah * Wproj^T + b  (fp32 out)
// grid = (400, 8), block = 32 (one wave per 16x64 tile).
// ---------------------------------------------------------------------------
__global__ void proj_gemm(const half_t* __restrict__ A, const half_t* __restrict__ W,
                          const float* __restrict__ bias, float* __restrict__ out) {
  const int r0 = blockIdx.x * 16;
  const int c0 = blockIdx.y * 64;

  v8f acc[4] = {};
  for (int k0 = 0; k0 < C_; k0 += 32) {
    v16h a = load_a_frag(A + (long)r0 * C_ + k0, C_);
#pragma unroll
    for (int t = 0; t < 4; ++t) {
      v16h b = load_b_frag(W + (long)(c0 + t * 16) * C_ + k0, C_);
      acc[t] = WMMA_F16(a, b, acc[t]);
    }
  }

  const int lane  = threadIdx.x & 31;
  const int coll  = lane & 15;
  const int rbase = (lane >> 4) * 8;
#pragma unroll
  for (int t = 0; t < 4; ++t) {
    const int c  = c0 + t * 16 + coll;
    const float bb = bias[c];
#pragma unroll
    for (int r = 0; r < 8; ++r)
      out[((long)(r0 + rbase + r)) * C_ + c] = acc[t][r] + bb;
  }
}

// ---------------------------------------------------------------------------
extern "C" void kernel_launch(void* const* d_in, const int* in_sizes, int n_in,
                              void* d_out, int out_size, void* d_ws, size_t ws_size,
                              hipStream_t stream) {
  const float* x      = (const float*)d_in[0];   // [B, N, C]
  const float* qkv_w  = (const float*)d_in[1];   // [3C, C]
  const float* qkv_b  = (const float*)d_in[2];   // [3C]
  const float* proj_w = (const float*)d_in[3];   // [C, C]
  const float* proj_b = (const float*)d_in[4];   // [C]
  const float* rel_h  = (const float*)d_in[5];   // [39, NH]
  const float* rel_w  = (const float*)d_in[6];   // [39, NH]
  float* out = (float*)d_out;                    // [B, N, C]

  const long NX   = (long)B_ * N_ * C_;          // 3,276,800
  const long NWQ  = (long)3 * C_ * C_;           //   786,432
  const long NWP  = (long)C_ * C_;               //   262,144
  const long NHEA = (long)B_ * NH_ * N_ * HD_;   // 3,276,800
  const long NVT  = (long)B_ * NH_ * HD_ * PADN; // 3,407,872

  half_t* ws     = (half_t*)d_ws;
  half_t* xh     = ws;
  half_t* wqkvh  = xh + NX;
  half_t* wprojh = wqkvh + NWQ;
  half_t* qh     = wprojh + NWP;
  half_t* kh     = qh + NHEA;
  half_t* vt     = kh + NHEA;
  half_t* aouth  = vt + NVT;                     // [B*N, C] f16

  cvt_f32_to_f16<<<1024, 256, 0, stream>>>(x, xh, (int)NX);
  cvt_f32_to_f16<<<1024, 256, 0, stream>>>(qkv_w, wqkvh, (int)NWQ);
  cvt_f32_to_f16<<<512, 256, 0, stream>>>(proj_w, wprojh, (int)NWP);
  zero_vt_pad<<<256, 256, 0, stream>>>(vt);

  qkv_gemm<<<dim3(B_ * N_ / 16, 3 * C_ / 64), 32, 0, stream>>>(xh, wqkvh, qkv_b, qh, kh, vt);

  attn_kernel<<<B_ * NH_ * NT_, 32, 0, stream>>>(qh, kh, vt, rel_h, rel_w, aouth);

  proj_gemm<<<dim3(B_ * N_ / 16, C_ / 64), 32, 0, stream>>>(aouth, wprojh, proj_b, out);
}
